// Attention_34557306864054
// MI455X (gfx1250) — matrix operand order repaired and checked by reading, not compiled, not run on hardware
//
#include <hip/hip_runtime.h>
#include <hip/hip_bf16.h>

// ---------------------------------------------------------------------------
// Attention block on gfx1250 (MI455X): bf16 WMMA GEMMs + flash attention.
//   B=4, N=2048, C=768, H=12, hd=64, SCALE=1/8
// Round 6: peel the final GEMM chunk so the A-prefetch/swap is unconditional
//          (kills the v_dual_cndmask select churn the round-5 guard caused),
//          and load each chunk's 8 B-fragments up-front for DS latency
//          distance.  Async-to-LDS staging + ASYNCcnt + barriers unchanged.
// ---------------------------------------------------------------------------

typedef __attribute__((ext_vector_type(16))) __bf16 v16bf;
typedef __attribute__((ext_vector_type(8)))  float  v8f;
typedef unsigned short ushort_t;

struct U4 { unsigned int x, y, z, w; };   // POD 16-byte chunk (union-safe)

#if defined(__AMDGCN__) && __has_builtin(__builtin_amdgcn_global_load_async_to_lds_b128)
#define ASYNC_LDS 1
typedef int v4i_ __attribute__((vector_size(16)));
typedef __attribute__((address_space(1))) v4i_* gptr128_t;   // global (__device__)
typedef __attribute__((address_space(3))) v4i_* lptr128_t;   // LDS (__shared__)
#else
#define ASYNC_LDS 0
#endif

__device__ __forceinline__ void wait_async_lds() {
#if ASYNC_LDS
#if __has_builtin(__builtin_amdgcn_s_wait_asynccnt)
  __builtin_amdgcn_s_wait_asynccnt(0);
#else
  asm volatile("s_wait_asynccnt 0" ::: "memory");
#endif
#endif
}

__device__ __forceinline__ ushort_t f32_to_bf16(float f) {
  unsigned int u = __float_as_uint(f);
  u += 0x7FFFu + ((u >> 16) & 1u);        // round-to-nearest-even
  return (ushort_t)(u >> 16);
}

// A-matrix 16x32 bf16 fragment (ISA 7.12.2): lane m=lane&15, group g=lane>>4.
// halves 0..7  -> K = k0 + 8*g .. +7   (VGPR0..3)
// halves 8..15 -> K = k0 + 16 + 8*g .. (VGPR4..7)
__device__ __forceinline__ v16bf load_frag_a(const ushort_t* __restrict__ base,
                                             int ld, int row0, int k0, int lane) {
  const int m = lane & 15, g = lane >> 4;
  const ushort_t* p = base + (size_t)(row0 + m) * ld + k0 + 8 * g;
  union { v16bf v; U4 q[2]; } u;
  u.q[0] = *reinterpret_cast<const U4*>(p);
  u.q[1] = *reinterpret_cast<const U4*>(p + 16);
  return u.v;
}

// B-matrix 32x16 bf16 fragment: lane n=lane&15 is the output column, g=lane>>4.
// halves 0..7 -> K = k0 + 8*g .. +7; halves 8..15 -> K = k0 + 16 + 8*g .. +7.
__device__ __forceinline__ v16bf load_frag_b(const ushort_t* __restrict__ base,
                                             int ld, int col0, int k0, int lane) {
  const int n = lane & 15, g = lane >> 4;
  const ushort_t* p = base + (size_t)(col0 + n) * ld + k0 + 8 * g;
  union { v16bf v; U4 q[2]; } u;
  u.q[0] = *reinterpret_cast<const U4*>(p);
  u.q[1] = *reinterpret_cast<const U4*>(p + 16);
  return u.v;
}

__device__ __forceinline__ v8f wmma_bf16(v16bf a, v16bf b, v8f c) {
  return __builtin_amdgcn_wmma_f32_16x16x32_bf16(false, a, false, b,
                                                 (short)0, c, false, false);
}

// ---------------------------------------------------------------------------
// Block-cooperative staging of a 64-col x 64-K bf16 weight chunk into LDS.
// Dest pitch = 72 halves (144 B): the 16 consuming lanes land on 16 distinct
// banks (36 dwords stride is unit-period-16 mod 64).  128 threads move
// 512 x 16 B = 8 KB, 4 transfers each.
// ---------------------------------------------------------------------------
#define LDS_B_PITCH 72

__device__ __forceinline__ void stage_b_chunk(const ushort_t* __restrict__ wb,
                                              int N0, int kc,
                                              ushort_t* __restrict__ dst,
                                              int tid) {
#pragma unroll
  for (int i = 0; i < 4; ++i) {
    const int idx = tid + 128 * i;            // 0..511
    const int row = idx >> 3, seg = idx & 7;  // row 0..63, 16B segment 0..7
    const ushort_t* src = wb + (size_t)(N0 + row) * 768 + kc + seg * 8;
    ushort_t* d = dst + row * LDS_B_PITCH + seg * 8;
#if ASYNC_LDS
    __builtin_amdgcn_global_load_async_to_lds_b128((gptr128_t)src, (lptr128_t)d,
                                                   0, 0);
#else
    *reinterpret_cast<U4*>(d) = *reinterpret_cast<const U4*>(src);
#endif
  }
}

// ---------------------------------------------------------------------------
// LDS-staged, double-buffered 32x64-per-wave GEMM body (block = 4 waves,
// 128M x 64N).  K = 768 -> 12 chunks of 64.  Main loop runs chunks 0..NC-2
// with unconditional next-chunk stage + A-prefetch; chunk NC-1 is peeled.
// ---------------------------------------------------------------------------
template <int K>
__device__ __forceinline__ void gemm_lds_body_32x64(
    const ushort_t* __restrict__ A, const ushort_t* __restrict__ Bm,
    int M0, int N0, int lane, int tid,
    ushort_t* __restrict__ lds0, ushort_t* __restrict__ lds1,
    v8f acc[2][4]) {
  constexpr int NC = K / 64;

  stage_b_chunk(Bm, N0, 0, lds0, tid);
  v16bf a[2][2];                              // [k-half][row-tile]
  a[0][0] = load_frag_a(A, K, M0,      0,  lane);
  a[0][1] = load_frag_a(A, K, M0 + 16, 0,  lane);
  a[1][0] = load_frag_a(A, K, M0,      32, lane);
  a[1][1] = load_frag_a(A, K, M0 + 16, 32, lane);
  wait_async_lds();
  __syncthreads();

#pragma unroll 2
  for (int c = 0; c < NC - 1; ++c) {
    const ushort_t* buf  = (c & 1) ? lds1 : lds0;
    ushort_t*       nbuf = (c & 1) ? lds0 : lds1;

    // unconditional stage + A-prefetch for chunk c+1 (overlaps WMMAs below)
    stage_b_chunk(Bm, N0, 64 * (c + 1), nbuf, tid);
    const int kn = 64 * (c + 1);
    v16bf na[2][2];
    na[0][0] = load_frag_a(A, K, M0,      kn,      lane);
    na[0][1] = load_frag_a(A, K, M0 + 16, kn,      lane);
    na[1][0] = load_frag_a(A, K, M0,      kn + 32, lane);
    na[1][1] = load_frag_a(A, K, M0 + 16, kn + 32, lane);

    // pull all 8 B-fragments of this chunk up-front (DS latency distance)
    v16bf bfr[2][4];
#pragma unroll
    for (int half = 0; half < 2; ++half)
#pragma unroll
      for (int t = 0; t < 4; ++t)
        bfr[half][t] = load_frag_b(buf, LDS_B_PITCH, 16 * t, 32 * half, lane);

#pragma unroll
    for (int half = 0; half < 2; ++half)
#pragma unroll
      for (int t = 0; t < 4; ++t) {
        acc[0][t] = wmma_bf16(a[half][0], bfr[half][t], acc[0][t]);
        acc[1][t] = wmma_bf16(a[half][1], bfr[half][t], acc[1][t]);
      }

    a[0][0] = na[0][0]; a[0][1] = na[0][1];
    a[1][0] = na[1][0]; a[1][1] = na[1][1];
    wait_async_lds();
    __syncthreads();
  }

  // peeled final chunk (NC-1)
  {
    const ushort_t* buf = ((NC - 1) & 1) ? lds1 : lds0;
    v16bf bfr[2][4];
#pragma unroll
    for (int half = 0; half < 2; ++half)
#pragma unroll
      for (int t = 0; t < 4; ++t)
        bfr[half][t] = load_frag_b(buf, LDS_B_PITCH, 16 * t, 32 * half, lane);
#pragma unroll
    for (int half = 0; half < 2; ++half)
#pragma unroll
      for (int t = 0; t < 4; ++t) {
        acc[0][t] = wmma_bf16(a[half][0], bfr[half][t], acc[0][t]);
        acc[1][t] = wmma_bf16(a[half][1], bfr[half][t], acc[1][t]);
      }
  }
}

// ---------------------------------------------------------------------------
// fp32 -> bf16 elementwise conversion
// ---------------------------------------------------------------------------
__global__ void cvt_f32_bf16_kernel(const float* __restrict__ src,
                                    ushort_t* __restrict__ dst, int n) {
  int i = blockIdx.x * blockDim.x + threadIdx.x;
  if (i < n) dst[i] = f32_to_bf16(src[i]);
}

// ---------------------------------------------------------------------------
// QKV GEMM:  qkv[m][o] = x[m][:] . Wqkv[o][:]  (+bias), m = B*N tokens.
// Epilogue scatters into flash layouts:
//   Q,K: [b,h,n,64]  (bf16, Q pre-scaled by 1/8, q-bias added)
//   V  : [b,h,64,n]  (bf16, transposed so PV B-frags are contiguous)
// ---------------------------------------------------------------------------
__global__ __launch_bounds__(128) void qkv_gemm_kernel(
    const ushort_t* __restrict__ xb, const ushort_t* __restrict__ wb,
    const float* __restrict__ q_bias, const float* __restrict__ v_bias,
    ushort_t* __restrict__ Qb, ushort_t* __restrict__ Kb,
    ushort_t* __restrict__ Vt) {
  __shared__ __align__(16) ushort_t lds_b[2][64 * LDS_B_PITCH];

  const int tid  = threadIdx.x;
  const int lane = tid & 31;
  const int wv   = tid >> 5;
  const int M0   = blockIdx.y * 128 + wv * 32;  // token tile (32 rows/wave)
  const int N0   = blockIdx.x * 64;             // output-feature tile (of 2304)

  v8f acc[2][4] = {};
  gemm_lds_body_32x64<768>(xb, wb, M0, N0, lane, tid, lds_b[0], lds_b[1], acc);

  const int n = lane & 15, g = lane >> 4;
#pragma unroll
  for (int rt = 0; rt < 2; ++rt) {
#pragma unroll
    for (int t = 0; t < 4; ++t) {
#pragma unroll
      for (int r = 0; r < 8; ++r) {
        const int token = M0 + 16 * rt + r + 8 * g;
        const int o     = N0 + 16 * t + n;
        const int sec   = o / 768;              // 0=q 1=k 2=v (uniform per tile)
        const int oc    = o % 768;
        const int h = oc >> 6, d = oc & 63;
        const int bidx = token >> 11, nt = token & 2047;
        float v = acc[rt][t][r];
        if (sec == 0) {
          v = (v + q_bias[oc]) * 0.125f;        // bias then SCALE (ref order)
          Qb[(((size_t)(bidx * 12 + h) * 2048 + nt) << 6) + d] = f32_to_bf16(v);
        } else if (sec == 1) {
          Kb[(((size_t)(bidx * 12 + h) * 2048 + nt) << 6) + d] = f32_to_bf16(v);
        } else {
          v += v_bias[oc];
          Vt[(((size_t)(bidx * 12 + h) * 64 + d) << 11) + nt] = f32_to_bf16(v);
        }
      }
    }
  }
}

// ---------------------------------------------------------------------------
// Flash attention: one wave per 16-row Q tile, streaming keys 64 at a time.
// Pipelined: V frags for block j and K frags for block j+1 are issued before
// the softmax VALU work; V lines for block j+1 are prefetched.
// Output: attn_out [B*N, C] bf16.
// ---------------------------------------------------------------------------
__global__ __launch_bounds__(256) void flash_attn_kernel(
    const ushort_t* __restrict__ Qb, const ushort_t* __restrict__ Kb,
    const ushort_t* __restrict__ Vt, ushort_t* __restrict__ attn_out) {
  __shared__ __align__(16) ushort_t lds_p[8 * 16 * 64];   // 2 KB per wave

  const int lane = threadIdx.x & 31;
  const int wv   = threadIdx.x >> 5;
  const int bh   = blockIdx.x;                 // 0..47
  const int b = bh / 12, h = bh % 12;
  const ushort_t* Qh = Qb + (size_t)bh * 2048 * 64;
  const ushort_t* Kh = Kb + (size_t)bh * 2048 * 64;
  const ushort_t* Vh = Vt + (size_t)bh * 64 * 2048;
  const int q0 = blockIdx.y * 128 + wv * 16;   // query tile base

  const v16bf aQ0 = load_frag_a(Qh, 64, q0, 0,  lane);
  const v16bf aQ1 = load_frag_a(Qh, 64, q0, 32, lane);

  v8f  o_acc[4] = {};
  float m_i[8], l_i[8];
#pragma unroll
  for (int r = 0; r < 8; ++r) { m_i[r] = -3.0e38f; l_i[r] = 0.0f; }

  ushort_t* myP = lds_p + wv * (16 * 64);
  const int n = lane & 15, g = lane >> 4;

  // preload K fragments for the first key block
  v16bf bk[4][2];
#pragma unroll
  for (int t = 0; t < 4; ++t) {
    bk[t][0] = load_frag_b(Kh, 64, 16 * t, 0,  lane);
    bk[t][1] = load_frag_b(Kh, 64, 16 * t, 32, lane);
  }

#pragma unroll 2
  for (int kb = 0; kb < 2048; kb += 64) {
    // ---- S = Q K^T for a 16x64 score tile ----
    v8f s[4] = {};
#pragma unroll
    for (int t = 0; t < 4; ++t) {
      s[t] = wmma_bf16(aQ0, bk[t][0], s[t]);
      s[t] = wmma_bf16(aQ1, bk[t][1], s[t]);
    }

    // ---- issue V loads for THIS block (consumed only after softmax) ----
    v16bf bv[4][2];
#pragma unroll
    for (int t = 0; t < 4; ++t) {
      bv[t][0] = load_frag_b(Vh, 2048, 16 * t, kb,      lane);
      bv[t][1] = load_frag_b(Vh, 2048, 16 * t, kb + 32, lane);
    }
    // ---- prefetch K fragments for the NEXT block; prefetch its V lines ----
    if (kb + 64 < 2048) {
#pragma unroll
      for (int t = 0; t < 4; ++t) {
        bk[t][0] = load_frag_b(Kh, 64, kb + 64 + 16 * t, 0,  lane);
        bk[t][1] = load_frag_b(Kh, 64, kb + 64 + 16 * t, 32, lane);
      }
#pragma unroll
      for (int t = 0; t < 4; ++t)   // one 128B line covers the 64-key block
        __builtin_prefetch(Vh + (size_t)(16 * t + n) * 2048 + kb + 64, 0, 3);
    }

    // ---- online softmax (rows per-VGPR; columns across 16 lanes) ----
    float mnew[8];
#pragma unroll
    for (int r = 0; r < 8; ++r) {
      mnew[r] = m_i[r];
#pragma unroll
      for (int t = 0; t < 4; ++t) mnew[r] = fmaxf(mnew[r], s[t][r]);
    }
#pragma unroll
    for (int r = 0; r < 8; ++r) {
#pragma unroll
      for (int off = 1; off < 16; off <<= 1)
        mnew[r] = fmaxf(mnew[r], __shfl_xor(mnew[r], off, 32));
    }
    float alpha[8], rsum[8];
#pragma unroll
    for (int r = 0; r < 8; ++r) { alpha[r] = __expf(m_i[r] - mnew[r]); rsum[r] = 0.0f; }
#pragma unroll
    for (int t = 0; t < 4; ++t) {
#pragma unroll
      for (int r = 0; r < 8; ++r) {
        float p = __expf(s[t][r] - mnew[r]);
        s[t][r] = p;
        rsum[r] += p;
      }
    }
#pragma unroll
    for (int r = 0; r < 8; ++r) {
#pragma unroll
      for (int off = 1; off < 16; off <<= 1)
        rsum[r] += __shfl_xor(rsum[r], off, 32);
      l_i[r] = l_i[r] * alpha[r] + rsum[r];
      m_i[r] = mnew[r];
    }
#pragma unroll
    for (int t = 0; t < 4; ++t)
#pragma unroll
      for (int r = 0; r < 8; ++r) o_acc[t][r] *= alpha[r];

    // ---- P: C-layout f32 -> LDS bf16 [16][64] -> A-layout frags ----
#pragma unroll
    for (int t = 0; t < 4; ++t)
#pragma unroll
      for (int r = 0; r < 8; ++r)
        myP[(r + 8 * g) * 64 + 16 * t + n] = f32_to_bf16(s[t][r]);
    asm volatile("s_wait_dscnt 0" ::: "memory");
    v16bf aP0 = load_frag_a(myP, 64, 0, 0,  lane);
    v16bf aP1 = load_frag_a(myP, 64, 0, 32, lane);

    // ---- O += P V ----
#pragma unroll
    for (int t = 0; t < 4; ++t) {
      o_acc[t] = wmma_bf16(aP0, bv[t][0], o_acc[t]);
      o_acc[t] = wmma_bf16(aP1, bv[t][1], o_acc[t]);
    }
  }

  // ---- normalize and write [B*N, C] bf16 ----
#pragma unroll
  for (int t = 0; t < 4; ++t)
#pragma unroll
    for (int r = 0; r < 8; ++r) {
      const int row = q0 + r + 8 * g;
      const int d   = 16 * t + n;
      float outv = o_acc[t][r] / l_i[r];
      attn_out[(size_t)(b * 2048 + row) * 768 + h * 64 + d] = f32_to_bf16(outv);
    }
}

// ---------------------------------------------------------------------------
// Output projection: out[m][o] = attn[m][:] . Wproj[o][:] + bias, f32 out.
// ---------------------------------------------------------------------------
__global__ __launch_bounds__(128) void proj_gemm_kernel(
    const ushort_t* __restrict__ ab, const ushort_t* __restrict__ wb,
    const float* __restrict__ bias, float* __restrict__ out) {
  __shared__ __align__(16) ushort_t lds_b[2][64 * LDS_B_PITCH];

  const int tid  = threadIdx.x;
  const int lane = tid & 31;
  const int wv   = tid >> 5;
  const int M0   = blockIdx.y * 128 + wv * 32;
  const int N0   = blockIdx.x * 64;

  v8f acc[2][4] = {};
  gemm_lds_body_32x64<768>(ab, wb, M0, N0, lane, tid, lds_b[0], lds_b[1], acc);

  const int n = lane & 15, g = lane >> 4;
#pragma unroll
  for (int rt = 0; rt < 2; ++rt)
#pragma unroll
    for (int t = 0; t < 4; ++t)
#pragma unroll
      for (int r = 0; r < 8; ++r) {
        const int token = M0 + 16 * rt + r + 8 * g;
        const int o     = N0 + 16 * t + n;
        out[(size_t)token * 768 + o] = acc[rt][t][r] + bias[o];
      }
}

// ---------------------------------------------------------------------------
// Host launcher
// ---------------------------------------------------------------------------
extern "C" void kernel_launch(void* const* d_in, const int* in_sizes, int n_in,
                              void* d_out, int out_size, void* d_ws, size_t ws_size,
                              hipStream_t stream) {
  (void)in_sizes; (void)n_in; (void)out_size; (void)ws_size;

  const float* x         = (const float*)d_in[0];   // [4,2048,768]
  const float* qkv_w     = (const float*)d_in[1];   // [2304,768]
  const float* q_bias    = (const float*)d_in[2];   // [768]
  const float* v_bias    = (const float*)d_in[3];   // [768]
  const float* proj_w    = (const float*)d_in[4];   // [768,768]
  const float* proj_bias = (const float*)d_in[5];   // [768]
  float* out = (float*)d_out;                       // [4,2048,768]

  constexpr size_t M  = 4 * 2048;      // 8192 tokens
  constexpr size_t C  = 768;
  constexpr size_t SZ_X    = M * C * 2;            // bf16 x
  constexpr size_t SZ_WQKV = 3 * C * C * 2;        // bf16 qkv weight
  constexpr size_t SZ_WPRJ = C * C * 2;            // bf16 proj weight
  constexpr size_t SZ_HEAD = M * C * 2;            // Q / K / V / attn (bf16)

  char* ws = (char*)d_ws;
  ushort_t* xb    = (ushort_t*)ws;              ws += SZ_X;
  ushort_t* wqkvb = (ushort_t*)ws;              ws += SZ_WQKV;
  ushort_t* wprjb = (ushort_t*)ws;              ws += SZ_WPRJ;
  ushort_t* Qb    = (ushort_t*)ws;              ws += SZ_HEAD;
  ushort_t* Kb    = (ushort_t*)ws;              ws += SZ_HEAD;
  ushort_t* Vt    = (ushort_t*)ws;              ws += SZ_HEAD;
  ushort_t* attnb = (ushort_t*)ws;              ws += SZ_HEAD;

  // 1) fp32 -> bf16 conversions
  {
    int n1 = (int)(M * C);
    cvt_f32_bf16_kernel<<<(n1 + 255) / 256, 256, 0, stream>>>(x, xb, n1);
    int n2 = (int)(3 * C * C);
    cvt_f32_bf16_kernel<<<(n2 + 255) / 256, 256, 0, stream>>>(qkv_w, wqkvb, n2);
    int n3 = (int)(C * C);
    cvt_f32_bf16_kernel<<<(n3 + 255) / 256, 256, 0, stream>>>(proj_w, wprjb, n3);
  }

  // 2) QKV GEMM + bias/scale + layout scatter   (2304/64=36, 8192/128=64)
  qkv_gemm_kernel<<<dim3(36, 64), 128, 0, stream>>>(xb, wqkvb, q_bias, v_bias,
                                                    Qb, Kb, Vt);

  // 3) Flash attention   (48 head-batches, 2048/128=16 query blocks)
  flash_attn_kernel<<<dim3(48, 16), 256, 0, stream>>>(Qb, Kb, Vt, attnb);

  // 4) Output projection  (768/64=12, 8192/128=64)
  proj_gemm_kernel<<<dim3(12, 64), 128, 0, stream>>>(attnb, wprjb, proj_bias, out);
}
